// AdaptiveFourierNeuralOperator_17540646437692
// MI455X (gfx1250) — matrix operand order repaired and checked by reading, not compiled, not run on hardware
//
#include <hip/hip_runtime.h>

typedef _Float16 half_t;
typedef _Float16 v16h __attribute__((ext_vector_type(16)));
typedef _Float16 v8h  __attribute__((ext_vector_type(8)));
typedef _Float16 v2h  __attribute__((ext_vector_type(2)));
typedef float    v8f  __attribute__((ext_vector_type(8)));

#define H_DIM 32
#define W_DIM 32
#define WF 17                 // W/2+1
#define C_DIM 768
#define BLK 192
#define NBLK 4
#define B_DIM 64
#define M_TOT (B_DIM * H_DIM * WF)      // 34816
#define FRAG_HALVES (NBLK * 12 * 6 * 32 * 16)  // 147456 per weight matrix

// Twiddles for 32-point FFT: cos/sin(2*pi*j/32), j=0..15
__constant__ float TWC[16] = {
    1.0f, 0.9807852804f, 0.9238795325f, 0.8314696123f,
    0.7071067812f, 0.5555702330f, 0.3826834324f, 0.1950903220f,
    0.0f, -0.1950903220f, -0.3826834324f, -0.5555702330f,
    -0.7071067812f, -0.8314696123f, -0.9238795325f, -0.9807852804f};
__constant__ float TWS[16] = {
    0.0f, 0.1950903220f, 0.3826834324f, 0.5555702330f,
    0.7071067812f, 0.8314696123f, 0.9238795325f, 0.9807852804f,
    1.0f, 0.9807852804f, 0.9238795325f, 0.8314696123f,
    0.7071067812f, 0.5555702330f, 0.3826834324f, 0.1950903220f};

// Fully register-resident 32-point complex FFT (radix-2 DIT, unnormalized).
template <bool INV>
__device__ __forceinline__ void fft32(float* re, float* im) {
#pragma unroll
  for (int i = 1; i < 32; i++) {
    int j = __brev((unsigned)i) >> 27;
    if (j > i) {
      float t = re[i]; re[i] = re[j]; re[j] = t;
      t = im[i]; im[i] = im[j]; im[j] = t;
    }
  }
#pragma unroll
  for (int s = 1; s <= 5; s++) {
    const int len = 1 << s, half = len >> 1;
#pragma unroll
    for (int i = 0; i < 32; i += len) {
#pragma unroll
      for (int j = 0; j < half; j++) {
        const int tw = j << (5 - s);
        const float wr = TWC[tw];
        const float wi = INV ? TWS[tw] : -TWS[tw];
        float xr = re[i + j + half], xi = im[i + j + half];
        float vr = xr * wr - xi * wi;
        float vi = xr * wi + xi * wr;
        float ur = re[i + j], ui = im[i + j];
        re[i + j] = ur + vr;        im[i + j] = ui + vi;
        re[i + j + half] = ur - vr; im[i + j + half] = ui - vi;
      }
    }
  }
}

// ---------------------------------------------------------------------------
// Weight swizzle: w[2][4][192][192] f32 -> WMMA B-fragment layout in f16.
// Frag layout: [blk][nT(12)][kS(6)][lane(32)][e(16)] halves.
// B element held by (lane,e): K = 32*kS + e + 16*(lane>=16), N = 16*nT + lane%16.
// ---------------------------------------------------------------------------
__global__ __launch_bounds__(256) void wswz_kernel(const float* __restrict__ w,
                                                   half_t* __restrict__ Wr,
                                                   half_t* __restrict__ Wi) {
  int idx = blockIdx.x * 256 + threadIdx.x;   // 0 .. 147455
  int e    = idx & 15;
  int lane = (idx >> 4) & 31;
  int kS   = (idx >> 9) % 6;
  int nT   = (idx / 3072) % 12;
  int blk  = idx / 36864;
  int d = 32 * kS + e + ((lane >= 16) ? 16 : 0);
  int n = 16 * nT + (lane & 15);
  int src = blk * (BLK * BLK) + d * BLK + n;
  Wr[idx] = (half_t)w[src];
  Wi[idx] = (half_t)w[NBLK * BLK * BLK + src];
}

// ---------------------------------------------------------------------------
// Forward rFFT2 (ortho). One wave per (batch, channel-pair); two real
// channels packed as one complex signal. Output Xr, Xi f16 [M_TOT][768].
// ---------------------------------------------------------------------------
__global__ __launch_bounds__(128) void fft_fwd_kernel(const float* __restrict__ x,
                                                      half_t* __restrict__ XR,
                                                      half_t* __restrict__ XI) {
  __shared__ float lds[4][32][33][2];  // [tile][h][w][re/im], 33 pad kills conflicts
  const int t = threadIdx.x;
  const int wg = blockIdx.x;            // 6144 WGs: 96 per batch x 64 batches
  const int b = wg / 96;
  const int cp0 = (wg % 96) * 4;        // first channel-pair of this WG
  const int c0g = cp0 * 2;              // first of 8 channels handled here

  // Cooperative coalesced load: 8 consecutive channels per position.
  const float* xb = x + (size_t)b * (H_DIM * W_DIM) * C_DIM;
  const int cOff = t & 7;
  const int pBase = t >> 3;             // 16 positions in flight
#pragma unroll 4
  for (int i = 0; i < 64; i++) {
    int p = i * 16 + pBase;
    float v = xb[(size_t)p * C_DIM + c0g + cOff];
    lds[cOff >> 1][p >> 5][p & 31][cOff & 1] = v;
  }
  __syncthreads();

  const int wave = t >> 5, lane = t & 31;
  float re[32], im[32];

  // Row FFT (along w); lane = h.
#pragma unroll
  for (int w = 0; w < 32; w++) { re[w] = lds[wave][lane][w][0]; im[w] = lds[wave][lane][w][1]; }
  fft32<false>(re, im);
#pragma unroll
  for (int w = 0; w < 32; w++) { lds[wave][lane][w][0] = re[w]; lds[wave][lane][w][1] = im[w]; }
  __syncthreads();

  // Column FFT (along h); lane = kw.
#pragma unroll
  for (int h = 0; h < 32; h++) { re[h] = lds[wave][h][lane][0]; im[h] = lds[wave][h][lane][1]; }
  fft32<false>(re, im);
#pragma unroll
  for (int h = 0; h < 32; h++) { lds[wave][h][lane][0] = re[h]; lds[wave][h][lane][1] = im[h]; }
  __syncthreads();

  // Unpack the two channels via conjugate symmetry; keep kw = 0..16.
  if (lane <= 16) {
    const int kw = lane, kwm = (32 - kw) & 31;
    const int cc0 = (cp0 + wave) * 2;
    const float sc = 1.0f / 64.0f;  // 0.5 unpack * 1/32 ortho
#pragma unroll
    for (int kh = 0; kh < 32; kh++) {
      const int khm = (32 - kh) & 31;
      float a = lds[wave][kh][kw][0],   bI = lds[wave][kh][kw][1];
      float c = lds[wave][khm][kwm][0], d  = lds[wave][khm][kwm][1];
      float x0r = (a + c) * sc, x0i = (bI - d) * sc;
      float x1r = (bI + d) * sc, x1i = (c - a) * sc;
      size_t m = ((size_t)b * H_DIM + kh) * WF + kw;
      v2h pr; pr[0] = (half_t)x0r; pr[1] = (half_t)x1r;
      v2h pi; pi[0] = (half_t)x0i; pi[1] = (half_t)x1i;
      *(v2h*)(XR + m * C_DIM + cc0) = pr;
      *(v2h*)(XI + m * C_DIM + cc0) = pi;
    }
  }
}

// ---------------------------------------------------------------------------
// Complex block-diagonal GEMM core. Each wave computes a 16(M) x 32(N) tile
// (two adjacent 16x16 n-tiles sharing the same A fragments):
//   O1 = act(A1@B1 + (NEG2 ? -1 : +1)*A2@B2 + bias1)
//   if DUAL: O2 = act(A1@B2 + A2@B1 + bias2)
// A*: f16 row-major [M_TOT][768]; B*f: fragment-swizzled f16 weights.
// K = 192 as 6 unrolled wmma_f32_16x16x32_f16 steps.
// WG = 8 waves = 4(M) x 2(N) -> 64M x 64N per WG; 192 % 64 == 0 so a WG
// never straddles a block boundary.
// ---------------------------------------------------------------------------
template <bool DUAL, bool RELU, bool NEG2>
__global__ __launch_bounds__(256) void gemm_cplx_kernel(
    const half_t* __restrict__ A1, const half_t* __restrict__ A2,
    const half_t* __restrict__ B1f, const half_t* __restrict__ B2f,
    const float* __restrict__ bias1, const float* __restrict__ bias2,
    half_t* __restrict__ O1, half_t* __restrict__ O2) {
  const int t = threadIdx.x, lane = t & 31, wave = t >> 5;
  const int wm = wave & 3, wn = wave >> 2;
  const int m0 = blockIdx.x * 64 + wm * 16;
  const int nG = blockIdx.y * 64 + wn * 32;      // base of this wave's 32 cols
  const int blk = nG / BLK, nB = nG % BLK, nT = nB >> 4;
  const int lo16 = lane & 15, hi = lane >> 4;
  const int row = m0 + lo16;

  const half_t* a1p = A1 + (size_t)row * C_DIM + blk * BLK;
  const half_t* a2p = A2 + (size_t)row * C_DIM + blk * BLK;
  // Fragments for n-tile nT and nT+1 (adjacent 16-column tiles).
  const half_t* b1p = B1f + ((size_t)(blk * 12 + nT) * 6) * 512 + lane * 16;
  const half_t* b2p = B2f + ((size_t)(blk * 12 + nT) * 6) * 512 + lane * 16;

  v8f acc1a = {}, acc1b = {};
  v8f acc2a = {}, acc2b = {};
#pragma unroll
  for (int s = 0; s < 6; s++) {
    // A fragment: lanes 0-15 = rows M, K = k0..k0+15 (+16 for hi lanes)
    v8h a1lo = *(const v8h*)(a1p + 32 * s + 16 * hi);
    v8h a1hi = *(const v8h*)(a1p + 32 * s + 8 + 16 * hi);
    v16h a1v = __builtin_shufflevector(a1lo, a1hi, 0, 1, 2, 3, 4, 5, 6, 7, 8, 9,
                                       10, 11, 12, 13, 14, 15);
    v8h a2lo = *(const v8h*)(a2p + 32 * s + 16 * hi);
    v8h a2hi = *(const v8h*)(a2p + 32 * s + 8 + 16 * hi);
    v16h a2v = __builtin_shufflevector(a2lo, a2hi, 0, 1, 2, 3, 4, 5, 6, 7, 8, 9,
                                       10, 11, 12, 13, 14, 15);
    v16h b1a = *(const v16h*)(b1p + s * 512);
    v16h b1b = *(const v16h*)(b1p + (6 + s) * 512);   // next n-tile
    v16h b2a = *(const v16h*)(b2p + s * 512);
    v16h b2b = *(const v16h*)(b2p + (6 + s) * 512);

    v16h a2x = NEG2 ? -a2v : a2v;   // f16 WMMA NEG bits are C-only -> negate in regs

    acc1a = __builtin_amdgcn_wmma_f32_16x16x32_f16(false, a1v, false, b1a,
                                                   (short)0, acc1a, false, false);
    acc1b = __builtin_amdgcn_wmma_f32_16x16x32_f16(false, a1v, false, b1b,
                                                   (short)0, acc1b, false, false);
    acc1a = __builtin_amdgcn_wmma_f32_16x16x32_f16(false, a2x, false, b2a,
                                                   (short)0, acc1a, false, false);
    acc1b = __builtin_amdgcn_wmma_f32_16x16x32_f16(false, a2x, false, b2b,
                                                   (short)0, acc1b, false, false);
    if (DUAL) {
      acc2a = __builtin_amdgcn_wmma_f32_16x16x32_f16(false, a1v, false, b2a,
                                                     (short)0, acc2a, false, false);
      acc2b = __builtin_amdgcn_wmma_f32_16x16x32_f16(false, a1v, false, b2b,
                                                     (short)0, acc2b, false, false);
      acc2a = __builtin_amdgcn_wmma_f32_16x16x32_f16(false, a2v, false, b1a,
                                                     (short)0, acc2a, false, false);
      acc2b = __builtin_amdgcn_wmma_f32_16x16x32_f16(false, a2v, false, b1b,
                                                     (short)0, acc2b, false, false);
    }
  }

  const float bv1a = bias1[blk * BLK + nB + lo16];
  const float bv1b = bias1[blk * BLK + nB + 16 + lo16];
  float bv2a = 0.0f, bv2b = 0.0f;
  if (DUAL) {
    bv2a = bias2[blk * BLK + nB + lo16];
    bv2b = bias2[blk * BLK + nB + 16 + lo16];
  }

#pragma unroll
  for (int v = 0; v < 8; v++) {
    size_t orow = (size_t)(m0 + v + 8 * hi) * C_DIM + nG + lo16;
    float o1a = acc1a[v] + bv1a;
    float o1b = acc1b[v] + bv1b;
    if (RELU) { o1a = fmaxf(o1a, 0.0f); o1b = fmaxf(o1b, 0.0f); }
    O1[orow]      = (half_t)o1a;
    O1[orow + 16] = (half_t)o1b;
    if (DUAL) {
      float o2a = acc2a[v] + bv2a;
      float o2b = acc2b[v] + bv2b;
      if (RELU) { o2a = fmaxf(o2a, 0.0f); o2b = fmaxf(o2b, 0.0f); }
      O2[orow]      = (half_t)o2a;
      O2[orow + 16] = (half_t)o2b;
    }
  }
}

// ---------------------------------------------------------------------------
// Inverse rFFT2 (ortho). Rebuild full spectrum (Hermitian mirror; kw=0,16
// columns explicitly symmetrized to match irfft2's real-part extraction),
// pack two channels, inverse FFT cols then rows, write f32 output.
// ---------------------------------------------------------------------------
__global__ __launch_bounds__(128) void fft_inv_kernel(const half_t* __restrict__ R2,
                                                      const half_t* __restrict__ I2,
                                                      float* __restrict__ out) {
  __shared__ float lds[4][32][33][2];
  const int t = threadIdx.x, wave = t >> 5, lane = t & 31;
  const int wg = blockIdx.x;
  const int b = wg / 96;
  const int cp = (wg % 96) * 4 + wave;
  const int c0 = cp * 2;

  if (lane <= 16) {
    const int kw = lane;
#pragma unroll
    for (int kh = 0; kh < 32; kh++) {
      size_t m = ((size_t)b * H_DIM + kh) * WF + kw;
      v2h rp = *(const v2h*)(R2 + m * C_DIM + c0);  // {r0, r1}
      v2h ip = *(const v2h*)(I2 + m * C_DIM + c0);  // {i0, i1}
      float r0 = (float)rp[0], r1 = (float)rp[1];
      float i0 = (float)ip[0], i1 = (float)ip[1];
      if (kw == 0 || kw == 16) {
        // Self-mirrored column: symmetrize S -> (S + conj(flip S))/2.
        int khm = (32 - kh) & 31;
        size_t mm = ((size_t)b * H_DIM + khm) * WF + kw;
        v2h rpm = *(const v2h*)(R2 + mm * C_DIM + c0);
        v2h ipm = *(const v2h*)(I2 + mm * C_DIM + c0);
        float s0r = 0.5f * (r0 + (float)rpm[0]);
        float s0i = 0.5f * (i0 - (float)ipm[0]);
        float s1r = 0.5f * (r1 + (float)rpm[1]);
        float s1i = 0.5f * (i1 - (float)ipm[1]);
        lds[wave][kh][kw][0] = s0r - s1i;
        lds[wave][kh][kw][1] = s0i + s1r;
      } else {
        lds[wave][kh][kw][0] = r0 - i1;
        lds[wave][kh][kw][1] = i0 + r1;
        int khm = (32 - kh) & 31, kwm = 32 - kw;
        lds[wave][khm][kwm][0] = r0 + i1;   // conj-mirror, packed
        lds[wave][khm][kwm][1] = r1 - i0;
      }
    }
  }
  __syncthreads();

  float re[32], im[32];
  // Inverse FFT along kh; lane = kw (all 32 columns).
#pragma unroll
  for (int kh = 0; kh < 32; kh++) { re[kh] = lds[wave][kh][lane][0]; im[kh] = lds[wave][kh][lane][1]; }
  fft32<true>(re, im);
#pragma unroll
  for (int kh = 0; kh < 32; kh++) { lds[wave][kh][lane][0] = re[kh]; lds[wave][kh][lane][1] = im[kh]; }
  __syncthreads();

  // Inverse FFT along kw; lane = h.
#pragma unroll
  for (int w = 0; w < 32; w++) { re[w] = lds[wave][lane][w][0]; im[w] = lds[wave][lane][w][1]; }
  fft32<true>(re, im);

  const float sc = 1.0f / 32.0f;  // ortho
  float* ob = out + ((size_t)b * (H_DIM * W_DIM) + lane * W_DIM) * C_DIM;
#pragma unroll
  for (int w = 0; w < 32; w++) {
    ob[(size_t)w * C_DIM + c0]     = re[w] * sc;  // channel c0
    ob[(size_t)w * C_DIM + c0 + 1] = im[w] * sc;  // channel c1
  }
}

// ---------------------------------------------------------------------------
extern "C" void kernel_launch(void* const* d_in, const int* in_sizes, int n_in,
                              void* d_out, int out_size, void* d_ws, size_t ws_size,
                              hipStream_t stream) {
  const float* x  = (const float*)d_in[0];
  const float* w1 = (const float*)d_in[1];
  const float* b1 = (const float*)d_in[2];
  const float* w2 = (const float*)d_in[3];
  const float* b2 = (const float*)d_in[4];
  float* out = (float*)d_out;

  half_t* ws = (half_t*)d_ws;
  const size_t SZ = (size_t)M_TOT * C_DIM;  // 26.7M halves per buffer
  half_t* XR = ws;
  half_t* XI = ws + SZ;
  half_t* R1 = ws + 2 * SZ;
  half_t* I1 = ws + 3 * SZ;
  half_t* W1R = ws + 4 * SZ;
  half_t* W1I = W1R + FRAG_HALVES;
  half_t* W2R = W1I + FRAG_HALVES;
  half_t* W2I = W2R + FRAG_HALVES;
  half_t* R2 = XR;  // safe alias: XR/XI dead after layer 1
  half_t* I2 = XI;

  // Weight fragment swizzle (f32 -> f16, WMMA B layout).
  wswz_kernel<<<FRAG_HALVES / 256, 256, 0, stream>>>(w1, W1R, W1I);
  wswz_kernel<<<FRAG_HALVES / 256, 256, 0, stream>>>(w2, W2R, W2I);

  // Forward rFFT2.
  fft_fwd_kernel<<<B_DIM * 96, 128, 0, stream>>>(x, XR, XI);

  dim3 g(M_TOT / 64, C_DIM / 64);  // (544, 12)
  // Layer 1: R1 = relu(Xr@W1r - Xi@W1i + b1r); I1 = relu(Xr@W1i + Xi@W1r + b1i)
  gemm_cplx_kernel<true, true, true><<<g, 256, 0, stream>>>(
      XR, XI, W1R, W1I, b1, b1 + NBLK * BLK, R1, I1);
  // Layer 2 real: r2 = R1@W2r - I1@W2i + b2r
  gemm_cplx_kernel<false, false, true><<<g, 256, 0, stream>>>(
      R1, I1, W2R, W2I, b2, nullptr, R2, nullptr);
  // Layer 2 imag (uses updated r2): i2 = r2@W2i + I1@W2r + b2i
  gemm_cplx_kernel<false, false, false><<<g, 256, 0, stream>>>(
      R2, I1, W2I, W2R, b2 + NBLK * BLK, nullptr, I2, nullptr);

  // Inverse rFFT2 -> output.
  fft_inv_kernel<<<B_DIM * 96, 128, 0, stream>>>(R2, I2, out);
}
